// QCO_2d_20590073217246
// MI455X (gfx1250) — matrix operand order repaired and checked by reading, not compiled, not run on hardware
//
#include <hip/hip_runtime.h>
#include <hip/hip_bf16.h>
#include <math.h>

// Problem constants (fixed by the reference)
#define NIMG 16
#define CCH  64
#define H    512
#define W    512
#define HW   (H*W)
#define LEV  16

typedef __attribute__((ext_vector_type(16))) _Float16 v16h;
typedef __attribute__((ext_vector_type(8)))  float    v8f;
typedef __attribute__((ext_vector_type(4)))  float    v4f;   // native vector: OK for NT builtins

// Workspace layout (float units). Total = (8192 + 16*262144)*4 B ~= 16.1 MB.
#define OFF_SUM  0        // 1024 floats : per-(n,c) sum of x
#define OFF_A    1024     // 1024 floats : normalized mean vector a[n,c]
#define OFF_MM   2048     // 32 uints    : ordered-encoded min ([n]) / max ([16+n])
#define OFF_STA  2080     // 4096 floats : sta accumulators (16 images x 256)
#define OFF_COS  8192     // 16*262144 floats : cos_sim map

__device__ __forceinline__ unsigned f2ord(float f) {
    unsigned u = __float_as_uint(f);
    return (u & 0x80000000u) ? ~u : (u | 0x80000000u);
}
__device__ __forceinline__ float ord2f(unsigned o) {
    unsigned u = (o & 0x80000000u) ? (o & 0x7FFFFFFFu) : ~o;
    return __uint_as_float(u);
}

// ---- K0: re-initialize accumulators every launch (graph-replay safe) ----
__global__ void k_init(float* ws) {
    int i = blockIdx.x * 256 + threadIdx.x;          // 16 blocks x 256 = 4096
    if (i < 1024) ws[OFF_SUM + i] = 0.0f;
    if (i < 4096) ws[OFF_STA + i] = 0.0f;
    if (i < 32) {
        unsigned* mm = (unsigned*)(ws + OFF_MM);
        mm[i] = (i < 16) ? 0xFFFFFFFFu : 0u;         // +inf-min / -inf-max (ordered enc)
    }
}

// ---- K1: per-(n,c) spatial sum (pass 1 over x, 1 GiB streamed, b128 NT loads) ----
__global__ void k_sum(const float* __restrict__ x, float* ws) {
    int nc    = blockIdx.x;                          // 0..1023 = n*64+c
    int chunk = blockIdx.y;                          // 0..15
    const v4f* p = (const v4f*)(x + (size_t)nc * HW) + (size_t)chunk * 4096;
    float s = 0.0f;
    #pragma unroll 4
    for (int i = 0; i < 16; ++i) {
        v4f v = __builtin_nontemporal_load(&p[i * 256 + threadIdx.x]);
        s += (v.x + v.y) + (v.z + v.w);
    }
    __shared__ float red[256];
    red[threadIdx.x] = s; __syncthreads();
    for (int st = 128; st > 0; st >>= 1) {
        if (threadIdx.x < st) red[threadIdx.x] += red[threadIdx.x + st];
        __syncthreads();
    }
    if (threadIdx.x == 0) atomicAdd(&ws[OFF_SUM + nc], red[0]);
}

// ---- K2: a[n,:] = normalize(mean vector) ----
__global__ void k_norm_mean(float* ws) {
    int n = blockIdx.x, c = threadIdx.x;             // 16 x 64
    float m = ws[OFF_SUM + n * 64 + c] * (1.0f / (float)HW);
    __shared__ float red[64];
    red[c] = m * m; __syncthreads();
    for (int st = 32; st > 0; st >>= 1) {
        if (c < st) red[c] += red[c + st];
        __syncthreads();
    }
    float nrm = sqrtf(red[0]);
    ws[OFF_A + n * 64 + c] = m / fmaxf(nrm, 1e-12f);
}

// ---- K3: cos_sim map + per-image min/max (pass 2 over x, b128 NT loads) ----
// Each thread computes 4 consecutive pixels; x reads are non-temporal so the
// 16 MB cos map survives in L2 for k_sta.
__global__ void k_cos(const float* __restrict__ x, float* ws) {
    int n  = blockIdx.y;
    int p4 = blockIdx.x * 256 + threadIdx.x;         // float4 pixel-group 0..65535
    const v4f*   xp = (const v4f*)(x + (size_t)n * CCH * HW) + p4;
    const float* a  = ws + OFF_A + n * 64;
    v4f dot = {0.f, 0.f, 0.f, 0.f};
    v4f nrm = {0.f, 0.f, 0.f, 0.f};
    #pragma unroll 4
    for (int c = 0; c < CCH; ++c) {
        v4f   v  = __builtin_nontemporal_load(&xp[(size_t)c * (HW / 4)]);
        float ac = a[c];
        dot.x = fmaf(ac, v.x, dot.x);  nrm.x = fmaf(v.x, v.x, nrm.x);
        dot.y = fmaf(ac, v.y, dot.y);  nrm.y = fmaf(v.y, v.y, nrm.y);
        dot.z = fmaf(ac, v.z, dot.z);  nrm.z = fmaf(v.z, v.z, nrm.z);
        dot.w = fmaf(ac, v.w, dot.w);  nrm.w = fmaf(v.w, v.w, nrm.w);
    }
    v4f cs;
    cs.x = dot.x / fmaxf(sqrtf(nrm.x), 1e-12f);
    cs.y = dot.y / fmaxf(sqrtf(nrm.y), 1e-12f);
    cs.z = dot.z / fmaxf(sqrtf(nrm.z), 1e-12f);
    cs.w = dot.w / fmaxf(sqrtf(nrm.w), 1e-12f);
    ((v4f*)(ws + OFF_COS + (size_t)n * HW))[p4] = cs;

    float lmn = fminf(fminf(cs.x, cs.y), fminf(cs.z, cs.w));
    float lmx = fmaxf(fmaxf(cs.x, cs.y), fmaxf(cs.z, cs.w));
    __shared__ float rmn[256], rmx[256];
    rmn[threadIdx.x] = lmn; rmx[threadIdx.x] = lmx; __syncthreads();
    for (int st = 128; st > 0; st >>= 1) {
        if (threadIdx.x < st) {
            rmn[threadIdx.x] = fminf(rmn[threadIdx.x], rmn[threadIdx.x + st]);
            rmx[threadIdx.x] = fmaxf(rmx[threadIdx.x], rmx[threadIdx.x + st]);
        }
        __syncthreads();
    }
    if (threadIdx.x == 0) {
        unsigned* mm = (unsigned*)(ws + OFF_MM);
        atomicMin(&mm[n],      f2ord(rmn[0]));
        atomicMax(&mm[16 + n], f2ord(rmx[0]));
    }
}

// ---- K4: sta[n,i,j] via v_wmma_f32_16x16x32_f16; one wave per (n, row h) ----
__global__ void __launch_bounds__(32) k_sta(float* ws) {
    int n = blockIdx.x;                              // image
    int h = blockIdx.y;                              // row 0..510 (row 511 contributes 0)
    int l = threadIdx.x;                             // lane 0..31
    const unsigned* mm = (const unsigned*)(ws + OFF_MM);
    float cmin = ord2f(mm[n]), cmax = ord2f(mm[16 + n]);
    int   m    = l & 15;                             // level index for this lane (A-row / B-col)
    float qlev = ((2.0f * (float)m + 1.0f) * (1.0f / 32.0f)) * (cmax - cmin) + cmin;

    const float* row0 = ws + OFF_COS + (size_t)n * HW + (size_t)h * W;
    const float* row1 = row0 + W;                    // h+1 < 512 guaranteed
    __shared__ float sA[32], sB[32];

    const int baseA = (l < 16) ? 0 : 8;              // A lane-half pixel base
    const int klo   = (l < 16) ? 0 : 16;             // unused (B also uses baseA)
    v8f acc = {};

    for (int cw = 0; cw < 16; ++cw) {
        int w0 = cw * 32;
        __syncthreads();
        sA[l] = row0[w0 + l];                        // cos at (h, w0+l)
        int wp = w0 + l + 1;                         // branchless clamp: value at
        wp = (wp > W - 1) ? (W - 1) : wp;            // the clamped slot is zeroed below
        sB[l] = row1[wp];
        __syncthreads();

        v16h a, b;
        // A: 16x32 (MxK) f16 layout — lane m: K {0..7,16..23}; lane m+16: K {8..15,24..31}
        #pragma unroll
        for (int e = 0; e < 16; ++e) {
            int pa  = baseA + e + ((e >= 8) ? 8 : 0);
            float d = sA[pa] - qlev;
            a[e] = (_Float16)__expf(-(1.0f / 64.0f) * d * d);
        }
        // B: 32x16 (KxN) f16 layout — lane m: col m, K {0..7,16..23}; lane m+16: col m, K {8..15,24..31}
        #pragma unroll
        for (int e = 0; e < 16; ++e) {
            int k   = baseA + e + ((e >= 8) ? 8 : 0);
            float d = sB[k] - qlev;
            float q = __expf(-(1.0f / 64.0f) * d * d);
            b[e] = (_Float16)((w0 + k + 1 < W) ? q : 0.0f);  // zero-pad right border
        }
        acc = __builtin_amdgcn_wmma_f32_16x16x32_f16(
            /*neg_a=*/false, a, /*neg_b=*/false, b,
            /*c_mod=*/(short)0, acc, /*reuse_a=*/false, /*reuse_b=*/false);
    }

    // C layout: VGPR r, lanes 0-15 -> (M=r, N=lane); lanes 16-31 -> (M=r+8, N=lane-16)
    float* sta = ws + OFF_STA + n * 256;
    #pragma unroll
    for (int r = 0; r < 8; ++r) {
        int M = r + ((l < 16) ? 0 : 8);
        atomicAdd(&sta[M * 16 + m], acc[r]);
    }
}

// ---- K5: normalize sta, compute q_levels, write (16,16,16,3) ----
__global__ void k_final(const float* __restrict__ ws, float* __restrict__ out) {
    int n = blockIdx.x, t = threadIdx.x;             // 16 x 256
    float s = ws[OFF_STA + n * 256 + t];
    __shared__ float red[256];
    red[t] = s; __syncthreads();
    for (int st = 128; st > 0; st >>= 1) {
        if (t < st) red[t] += red[t + st];
        __syncthreads();
    }
    float tot = red[0];
    const unsigned* mm = (const unsigned*)(ws + OFF_MM);
    float cmin = ord2f(mm[n]), cmax = ord2f(mm[16 + n]);
    int i = t >> 4, j = t & 15;
    float qi = ((2.0f * (float)i + 1.0f) / 32.0f) * (cmax - cmin) + cmin;
    float qj = ((2.0f * (float)j + 1.0f) / 32.0f) * (cmax - cmin) + cmin;
    size_t o = ((size_t)(n * 16 + i) * 16 + (size_t)j) * 3;
    out[o + 0] = qj;        // qh[n,i,j] = q_levels[n, j]
    out[o + 1] = qi;        // qw[n,i,j] = q_levels[n, i]
    out[o + 2] = s / tot;   // normalized co-occurrence
}

extern "C" void kernel_launch(void* const* d_in, const int* in_sizes, int n_in,
                              void* d_out, int out_size, void* d_ws, size_t ws_size,
                              hipStream_t stream) {
    const float* x   = (const float*)d_in[0];
    float*       out = (float*)d_out;
    float*       ws  = (float*)d_ws;   // needs ~16.1 MB

    k_init     <<<16,              256, 0, stream>>>(ws);
    k_sum      <<<dim3(1024, 16),  256, 0, stream>>>(x, ws);
    k_norm_mean<<<16,               64, 0, stream>>>(ws);
    k_cos      <<<dim3(256, 16),   256, 0, stream>>>(x, ws);
    k_sta      <<<dim3(16, 511),    32, 0, stream>>>(ws);
    k_final    <<<16,              256, 0, stream>>>(ws, out);
}
